// GraphSAGE_DGL_19851338842541
// MI455X (gfx1250) — matrix operand order repaired and checked by reading, not compiled, not run on hardware
//
#include <hip/hip_runtime.h>
#include <hip/hip_bf16.h>
#include <stdint.h>

#define N_NODES 50000
#define N_EDGES 800000
#define K_DIM   128      // input dim of every layer (IN == HID == 128)

typedef __attribute__((ext_vector_type(16))) __bf16 v16bf;
typedef __attribute__((ext_vector_type(8)))  float  v8f;

union FragAB {
  v16bf        v;
  unsigned int u[8];
  uint4        q[2];
};

__device__ __forceinline__ unsigned int f32_hi_bits(float f) {
  union { float f; unsigned int u; } c; c.f = f; return c.u >> 16;
}
__device__ __forceinline__ float hi_to_f32(unsigned short h) {
  union { float f; unsigned int u; } c; c.u = ((unsigned int)h) << 16; return c.f;
}

// ---------------------------------------------------------------- utilities
__global__ void fill_zero_f32(float* __restrict__ p, int n) {
  int i = blockIdx.x * blockDim.x + threadIdx.x;
  if (i < n) p[i] = 0.0f;
}

__global__ void degree_kernel(const int* __restrict__ dst, float* __restrict__ deg, int e) {
  int i = blockIdx.x * blockDim.x + threadIdx.x;
  if (i < e) atomicAdd(&deg[dst[i]], 1.0f);
}

__global__ void invdeg_kernel(float* __restrict__ deg, int n) {
  int i = blockIdx.x * blockDim.x + threadIdx.x;
  if (i < n) deg[i] = 1.0f / fmaxf(deg[i], 1.0f);
}

// one wave per edge; lane handles 4 consecutive features (32 lanes * 4 = 128)
__global__ void scatter_kernel(const float* __restrict__ h, const int* __restrict__ src,
                               const int* __restrict__ dst, float* __restrict__ agg, int e) {
  int edge = (blockIdx.x * blockDim.x + threadIdx.x) >> 5;
  int lane = threadIdx.x & 31;
  if (edge >= e) return;
  int s = src[edge];
  int d = dst[edge];
  float4 v = ((const float4*)(h + (size_t)s * K_DIM))[lane];
  float* a = agg + (size_t)d * K_DIM + lane * 4;
  atomicAdd(a + 0, v.x);
  atomicAdd(a + 1, v.y);
  atomicAdd(a + 2, v.z);
  atomicAdd(a + 3, v.w);
}

// ------------------------------------------------- weight packing (bf16 hi/lo split)
// Packs W[K][D] (row-major f32) into B-fragment order for v_wmma_f32_16x16x32_bf16:
// idx = ((kt*NT + nt)*32 + lane)*16 + e ; lane -> col = nt*16 + (lane&15),
// k = kt*32 + (e<8 ? 8*(lane>>4)+e : 16+8*(lane>>4)+(e-8))   (ISA 16-bit B 32x16 layout)
__global__ void pack_weight_bf16(const float* __restrict__ W,
                                 unsigned short* __restrict__ whi,
                                 unsigned short* __restrict__ wlo, int K, int D) {
  int idx = blockIdx.x * blockDim.x + threadIdx.x;
  if (idx >= K * D) return;
  int e    = idx & 15;
  int lane = (idx >> 4) & 31;
  int tile = idx >> 9;
  int NT   = D >> 4;
  int nt   = tile % NT;
  int kt   = tile / NT;
  int k    = kt * 32 + ((e < 8) ? (8 * (lane >> 4) + e) : (16 + 8 * (lane >> 4) + (e - 8)));
  int col  = nt * 16 + (lane & 15);
  float w  = W[k * D + col];
  unsigned short hb = (unsigned short)f32_hi_bits(w);
  float lo = w - hi_to_f32(hb);
  unsigned short lb = (unsigned short)f32_hi_bits(lo);
  whi[idx] = hb;
  wlo[idx] = lb;
}

// ------------------------------------------------- fused SAGE layer GEMM
// out[n, :] = act( h[n,:] @ Ws + (agg[n,:]*invdeg[n]) @ Wn + b )
// One wave per 16x16 output tile; split-bf16 (hi*hi + lo*hi + hi*lo) for ~fp32 accuracy.
template <int DOUT, bool RELU>
__global__ void sage_gemm(const float* __restrict__ h, const float* __restrict__ agg,
                          const float* __restrict__ invdeg,
                          const unsigned short* __restrict__ ws_hi,
                          const unsigned short* __restrict__ ws_lo,
                          const unsigned short* __restrict__ wn_hi,
                          const unsigned short* __restrict__ wn_lo,
                          const float* __restrict__ bias, float* __restrict__ out) {
  constexpr int NT = DOUT / 16;
  __shared__ unsigned short sAh_hi[16 * K_DIM];
  __shared__ unsigned short sAh_lo[16 * K_DIM];
  __shared__ unsigned short sAm_hi[16 * K_DIM];
  __shared__ unsigned short sAm_lo[16 * K_DIM];

  const int tid  = threadIdx.x;
  const int row0 = blockIdx.x * 16;

  // stage 16 rows of self features and mean-aggregated features as bf16 hi/lo
  for (int i = tid; i < 16 * K_DIM; i += blockDim.x) {
    int r = i >> 7, c = i & (K_DIM - 1);
    int row = row0 + r;
    float hv = 0.0f, mv = 0.0f;
    if (row < N_NODES) {
      hv = h[(size_t)row * K_DIM + c];
      mv = agg[(size_t)row * K_DIM + c] * invdeg[row];
    }
    unsigned short hh = (unsigned short)f32_hi_bits(hv);
    sAh_hi[i] = hh;
    sAh_lo[i] = (unsigned short)f32_hi_bits(hv - hi_to_f32(hh));
    unsigned short mh = (unsigned short)f32_hi_bits(mv);
    sAm_hi[i] = mh;
    sAm_lo[i] = (unsigned short)f32_hi_bits(mv - hi_to_f32(mh));
  }
  __syncthreads();

  const int lane = tid & 31;
  const int wave = tid >> 5;        // output 16-col tile index
  const int half = lane >> 4;
  const int mrow = lane & 15;

  v8f acc = {};

  // self term: h @ Ws
  #pragma unroll
  for (int kt = 0; kt < K_DIM / 32; ++kt) {
    FragAB ahi, alo, bhi, blo;
    #pragma unroll
    for (int d = 0; d < 8; ++d) {
      int ks  = (d < 4) ? (8 * half + 2 * d) : (16 + 8 * half + 2 * (d - 4));
      int off = mrow * K_DIM + kt * 32 + ks;
      ahi.u[d] = *(const unsigned int*)&sAh_hi[off];
      alo.u[d] = *(const unsigned int*)&sAh_lo[off];
    }
    const uint4* ph = (const uint4*)&ws_hi[((kt * NT + wave) * 32 + lane) * 16];
    const uint4* pl = (const uint4*)&ws_lo[((kt * NT + wave) * 32 + lane) * 16];
    bhi.q[0] = ph[0]; bhi.q[1] = ph[1];
    blo.q[0] = pl[0]; blo.q[1] = pl[1];
    acc = __builtin_amdgcn_wmma_f32_16x16x32_bf16(false, ahi.v, false, bhi.v, (short)0, acc, false, false);
    acc = __builtin_amdgcn_wmma_f32_16x16x32_bf16(false, alo.v, false, bhi.v, (short)0, acc, false, false);
    acc = __builtin_amdgcn_wmma_f32_16x16x32_bf16(false, ahi.v, false, blo.v, (short)0, acc, false, false);
  }

  // neighbor term: mean @ Wn
  #pragma unroll
  for (int kt = 0; kt < K_DIM / 32; ++kt) {
    FragAB ahi, alo, bhi, blo;
    #pragma unroll
    for (int d = 0; d < 8; ++d) {
      int ks  = (d < 4) ? (8 * half + 2 * d) : (16 + 8 * half + 2 * (d - 4));
      int off = mrow * K_DIM + kt * 32 + ks;
      ahi.u[d] = *(const unsigned int*)&sAm_hi[off];
      alo.u[d] = *(const unsigned int*)&sAm_lo[off];
    }
    const uint4* ph = (const uint4*)&wn_hi[((kt * NT + wave) * 32 + lane) * 16];
    const uint4* pl = (const uint4*)&wn_lo[((kt * NT + wave) * 32 + lane) * 16];
    bhi.q[0] = ph[0]; bhi.q[1] = ph[1];
    blo.q[0] = pl[0]; blo.q[1] = pl[1];
    acc = __builtin_amdgcn_wmma_f32_16x16x32_bf16(false, ahi.v, false, bhi.v, (short)0, acc, false, false);
    acc = __builtin_amdgcn_wmma_f32_16x16x32_bf16(false, alo.v, false, bhi.v, (short)0, acc, false, false);
    acc = __builtin_amdgcn_wmma_f32_16x16x32_bf16(false, ahi.v, false, blo.v, (short)0, acc, false, false);
  }

  // epilogue: C/D layout -> VGPR r: lanes 0-15 M=r, lanes 16-31 M=8+r; N=lane&15
  const int ncol = wave * 16 + (lane & 15);
  const float bv = bias[ncol];
  #pragma unroll
  for (int r = 0; r < 8; ++r) {
    int row = row0 + r + 8 * half;
    if (row < N_NODES) {
      float v = acc[r] + bv;
      if (RELU) v = fmaxf(v, 0.0f);
      out[(size_t)row * DOUT + ncol] = v;
    }
  }
}

// ---------------------------------------------------------------- host
extern "C" void kernel_launch(void* const* d_in, const int* in_sizes, int n_in,
                              void* d_out, int out_size, void* d_ws, size_t ws_size,
                              hipStream_t stream) {
  (void)in_sizes; (void)n_in; (void)out_size; (void)ws_size;

  const float* x   = (const float*)d_in[0];
  const int*   src = (const int*)d_in[1];
  const int*   dst = (const int*)d_in[2];
  const float* Ws0 = (const float*)d_in[3];
  const float* Wn0 = (const float*)d_in[4];
  const float* b0  = (const float*)d_in[5];
  const float* Ws1 = (const float*)d_in[6];
  const float* Wn1 = (const float*)d_in[7];
  const float* b1  = (const float*)d_in[8];
  const float* Ws2 = (const float*)d_in[9];
  const float* Wn2 = (const float*)d_in[10];
  const float* b2  = (const float*)d_in[11];
  float* out = (float*)d_out;

  char*  ws  = (char*)d_ws;
  size_t off = 0;
  auto alloc = [&](size_t bytes) -> void* {
    void* p = ws + off;
    off = (off + bytes + 255) & ~(size_t)255;
    return p;
  };

  float* invdeg = (float*)alloc((size_t)N_NODES * 4);
  float* agg    = (float*)alloc((size_t)N_NODES * K_DIM * 4);
  float* h1     = (float*)alloc((size_t)N_NODES * K_DIM * 4);
  float* h2     = (float*)alloc((size_t)N_NODES * K_DIM * 4);

  unsigned short* s0h = (unsigned short*)alloc(128 * 128 * 2);
  unsigned short* s0l = (unsigned short*)alloc(128 * 128 * 2);
  unsigned short* n0h = (unsigned short*)alloc(128 * 128 * 2);
  unsigned short* n0l = (unsigned short*)alloc(128 * 128 * 2);
  unsigned short* s1h = (unsigned short*)alloc(128 * 128 * 2);
  unsigned short* s1l = (unsigned short*)alloc(128 * 128 * 2);
  unsigned short* n1h = (unsigned short*)alloc(128 * 128 * 2);
  unsigned short* n1l = (unsigned short*)alloc(128 * 128 * 2);
  unsigned short* s2h = (unsigned short*)alloc(128 * 64 * 2);
  unsigned short* s2l = (unsigned short*)alloc(128 * 64 * 2);
  unsigned short* n2h = (unsigned short*)alloc(128 * 64 * 2);
  unsigned short* n2l = (unsigned short*)alloc(128 * 64 * 2);

  const int feat_elems = N_NODES * K_DIM;   // 6.4M
  const int TB = 256;

  // degrees -> inverse degrees
  fill_zero_f32<<<(N_NODES + TB - 1) / TB, TB, 0, stream>>>(invdeg, N_NODES);
  degree_kernel<<<(N_EDGES + TB - 1) / TB, TB, 0, stream>>>(dst, invdeg, N_EDGES);
  invdeg_kernel<<<(N_NODES + TB - 1) / TB, TB, 0, stream>>>(invdeg, N_NODES);

  // pack weights into WMMA B-fragment bf16 hi/lo planes
  pack_weight_bf16<<<(128 * 128 + TB - 1) / TB, TB, 0, stream>>>(Ws0, s0h, s0l, 128, 128);
  pack_weight_bf16<<<(128 * 128 + TB - 1) / TB, TB, 0, stream>>>(Wn0, n0h, n0l, 128, 128);
  pack_weight_bf16<<<(128 * 128 + TB - 1) / TB, TB, 0, stream>>>(Ws1, s1h, s1l, 128, 128);
  pack_weight_bf16<<<(128 * 128 + TB - 1) / TB, TB, 0, stream>>>(Wn1, n1h, n1l, 128, 128);
  pack_weight_bf16<<<(128 * 64 + TB - 1) / TB, TB, 0, stream>>>(Ws2, s2h, s2l, 128, 64);
  pack_weight_bf16<<<(128 * 64 + TB - 1) / TB, TB, 0, stream>>>(Wn2, n2h, n2l, 128, 64);

  const int scatter_grid = (N_EDGES * 32 + TB - 1) / TB;  // one wave per edge
  const int gemm_grid    = (N_NODES + 15) / 16;           // 3125 (exact)

  // layer 0
  fill_zero_f32<<<(feat_elems + TB - 1) / TB, TB, 0, stream>>>(agg, feat_elems);
  scatter_kernel<<<scatter_grid, TB, 0, stream>>>(x, src, dst, agg, N_EDGES);
  sage_gemm<128, true><<<gemm_grid, 256, 0, stream>>>(x, agg, invdeg, s0h, s0l, n0h, n0l, b0, h1);

  // layer 1
  fill_zero_f32<<<(feat_elems + TB - 1) / TB, TB, 0, stream>>>(agg, feat_elems);
  scatter_kernel<<<scatter_grid, TB, 0, stream>>>(h1, src, dst, agg, N_EDGES);
  sage_gemm<128, true><<<gemm_grid, 256, 0, stream>>>(h1, agg, invdeg, s1h, s1l, n1h, n1l, b1, h2);

  // layer 2 (no relu, DOUT=64)
  fill_zero_f32<<<(feat_elems + TB - 1) / TB, TB, 0, stream>>>(agg, feat_elems);
  scatter_kernel<<<scatter_grid, TB, 0, stream>>>(h2, src, dst, agg, N_EDGES);
  sage_gemm<64, false><<<gemm_grid, 128, 0, stream>>>(h2, agg, invdeg, s2h, s2l, n2h, n2l, b2, out);
}
